// PNA_Net_68521908240399
// MI455X (gfx1250) — matrix operand order repaired and checked by reading, not compiled, not run on hardware
//
#include <hip/hip_runtime.h>
#include <hip/hip_bf16.h>
#include <math.h>

// ---------------- problem constants (match reference) ----------------
#define NN    50000
#define EE    800000
#define FF    80
#define LL    4
#define NGG   1024
#define NFEAT 9
#define VOCAB 119
#define KIN   1040      // 13*F
#define KPAD  1056      // padded to multiple of 32
#define KEDGE 160       // 2*F
#define EPITCH 168      // LDS row pitch for preWt tile (16B aligned, conflict-spread)
#define NPITCH 40       // LDS row pitch for node-B chunk

typedef __attribute__((ext_vector_type(16))) __bf16 bf16x16;
typedef __attribute__((ext_vector_type(8)))  __bf16 bf16x8;
typedef __attribute__((ext_vector_type(8)))  float  f32x8;

__device__ __forceinline__ bf16x16 cat8(bf16x8 lo, bf16x8 hi) {
  bf16x16 r;
#pragma unroll
  for (int j = 0; j < 8; ++j) { r[j] = lo[j]; r[j + 8] = hi[j]; }
  return r;
}

// CDNA5 float atomics (no-return form, tracked via STOREcnt)
__device__ __forceinline__ void atomic_min_f32(float* p, float v) {
  asm volatile("global_atomic_min_num_f32 %0, %1, off" : : "v"(p), "v"(v) : "memory");
}
__device__ __forceinline__ void atomic_max_f32(float* p, float v) {
  asm volatile("global_atomic_max_num_f32 %0, %1, off" : : "v"(p), "v"(v) : "memory");
}

// ---------------- elementwise / prep kernels ----------------

__global__ void k_fill_f32(float* p, int n, float v) {
  int i = blockIdx.x * blockDim.x + threadIdx.x;
  if (i < n) p[i] = v;
}

__global__ void k_init_seg(float* s1, float* s2, float* mn, float* mx, int n) {
  int i = blockIdx.x * blockDim.x + threadIdx.x;
  if (i < n) { s1[i] = 0.f; s2[i] = 0.f; mn[i] = __builtin_inff(); mx[i] = -__builtin_inff(); }
}

__global__ void k_deg(const int* __restrict__ dst, float* __restrict__ deg, int e) {
  int i = blockIdx.x * blockDim.x + threadIdx.x;
  if (i < e) atomicAdd(&deg[dst[i]], 1.0f);
}

__global__ void k_atom_encoder(const int* __restrict__ xf, const float* __restrict__ emb,
                               float* __restrict__ h, __bf16* __restrict__ hbf, int n) {
  int i = blockIdx.x * blockDim.x + threadIdx.x;
  if (i >= n * FF) return;
  int node = i / FF, f = i % FF;
  float acc = 0.f;
#pragma unroll
  for (int c = 0; c < NFEAT; ++c) {
    int v = xf[node * NFEAT + c];
    acc += emb[((size_t)c * VOCAB + v) * FF + f];
  }
  h[i] = acc;
  hbf[i] = (__bf16)acc;
}

// transpose pre_W[l] (160x80) -> preWt[l] (80x160) bf16
__global__ void k_prep_preWt(const float* __restrict__ preW, __bf16* __restrict__ preWt) {
  int i = blockIdx.x * blockDim.x + threadIdx.x;
  if (i >= LL * FF * KEDGE) return;
  int l = i / (FF * KEDGE), rem = i % (FF * KEDGE);
  int f = rem / KEDGE, k = rem % KEDGE;
  preWt[i] = (__bf16)preW[((size_t)l * KEDGE + k) * FF + f];
}

// combWt[l][f][k] = (post_W[l] @ lin_W[l])^T, bf16, K padded to 1056
__global__ void k_prep_comb(const float* __restrict__ postW, const float* __restrict__ linW,
                            __bf16* __restrict__ combWt) {
  int i = blockIdx.x * blockDim.x + threadIdx.x;
  if (i >= LL * FF * KPAD) return;
  int l = i / (FF * KPAD), rem = i % (FF * KPAD);
  int f = rem / KPAD, k = rem % KPAD;
  float acc = 0.f;
  if (k < KIN) {
    const float* pw = postW + ((size_t)l * KIN + k) * FF;
    const float* lw = linW + (size_t)l * FF * FF;
#pragma unroll 8
    for (int t = 0; t < FF; ++t) acc += pw[t] * lw[t * FF + f];
  }
  combWt[i] = (__bf16)acc;
}

// combB[l][f] = post_b[l] @ lin_W[l] + lin_b[l]
__global__ void k_prep_combB(const float* __restrict__ postb, const float* __restrict__ linW,
                             const float* __restrict__ linb, float* __restrict__ combB) {
  int i = blockIdx.x * blockDim.x + threadIdx.x;
  if (i >= LL * FF) return;
  int l = i / FF, f = i % FF;
  float acc = linb[i];
  const float* pb = postb + (size_t)l * FF;
  const float* lw = linW + (size_t)l * FF * FF;
#pragma unroll 8
  for (int t = 0; t < FF; ++t) acc += pb[t] * lw[t * FF + f];
  combB[i] = acc;
}

// build X[n, 0:1056] bf16 = cat([h, agg, agg*sc, agg/sc], pad)
__global__ void k_build_x(const float* __restrict__ h, const float* __restrict__ deg,
                          const float* __restrict__ avgp,
                          const float* __restrict__ s1, const float* __restrict__ s2,
                          const float* __restrict__ smn, const float* __restrict__ smx,
                          __bf16* __restrict__ X, int n) {
  int i = blockIdx.x * blockDim.x + threadIdx.x;
  if (i >= n * FF) return;
  int node = i / FF, f = i % FF;
  float dg = deg[node];
  float degc = fmaxf(dg, 1.f);
  bool he = dg > 0.f;
  float sc = logf(degc + 1.f) / avgp[0];
  float mean = s1[i] / degc;
  float var = s2[i] / degc - mean * mean;
  float stdv = sqrtf(fmaxf(var, 0.f) + 1e-5f);
  float mn = he ? smn[i] : 0.f;
  float mx = he ? smx[i] : 0.f;
  size_t b = (size_t)node * KPAD;
  X[b + f] = (__bf16)h[i];
  float agg[4] = {mean, mn, mx, stdv};
#pragma unroll
  for (int t = 0; t < 4; ++t) {
    X[b + 80 + t * 80 + f]  = (__bf16)agg[t];
    X[b + 400 + t * 80 + f] = (__bf16)(agg[t] * sc);
    X[b + 720 + t * 80 + f] = (__bf16)(agg[t] / sc);
  }
  if (f < KPAD - KIN) X[b + KIN + f] = (__bf16)0.f;
}

__global__ void k_bn_relu_res(const float* __restrict__ out1,
                              const float* __restrict__ gamma, const float* __restrict__ beta,
                              const float* __restrict__ mean, const float* __restrict__ var,
                              float* __restrict__ h, __bf16* __restrict__ hbf, int n) {
  int i = blockIdx.x * blockDim.x + threadIdx.x;
  if (i >= n * FF) return;
  int f = i % FF;
  float o = out1[i];
  o = (o - mean[f]) * (gamma[f] * rsqrtf(var[f] + 1e-5f)) + beta[f];
  float hn = h[i] + fmaxf(o, 0.f);
  h[i] = hn;
  hbf[i] = (__bf16)hn;
}

__global__ void k_pool_accum(const float* __restrict__ h, const int* __restrict__ batch,
                             float* __restrict__ gsum, float* __restrict__ gcnt, int n) {
  int i = blockIdx.x * blockDim.x + threadIdx.x;
  if (i >= n * FF) return;
  int node = i / FF, f = i % FF;
  int g = batch[node];
  atomicAdd(&gsum[(size_t)g * FF + f], h[i]);
  if (f == 0) atomicAdd(&gcnt[g], 1.0f);
}

__global__ void k_final(const float* __restrict__ gsum, const float* __restrict__ gcnt,
                        const float* __restrict__ mlpW, const float* __restrict__ mlpb,
                        float* __restrict__ out, int ng) {
  int g = blockIdx.x * blockDim.x + threadIdx.x;
  if (g >= ng) return;
  float c = fmaxf(gcnt[g], 1.f);
  float acc = mlpb[0];
#pragma unroll 8
  for (int f = 0; f < FF; ++f) acc += (gsum[(size_t)g * FF + f] / c) * mlpW[f];
  out[g] = acc;
}

// ---------------- WMMA kernels ----------------
// Edge-message GEMM + fused segment reductions.
// preWt (80x160 bf16, 25.6KB) staged once per block into LDS (rows padded to
// EPITCH for 16B alignment + bank spread). Each wave owns a 16-edge tile:
// A = cat(h[dst],h[src]) via two b128 loads per K-chunk (runs never cross the
// dst/src split at 80), B from ds_load_b128; 5x5 v_wmma_f32_16x16x32_bf16;
// epilogue does atomic add / add(m^2) / min_num / max_num per node feature.
__global__ void k_edge_wmma(const __bf16* __restrict__ hbf,
                            const __bf16* __restrict__ preWt,   // [80][160] bf16
                            const float* __restrict__ preb,     // [80]
                            const int* __restrict__ srcI, const int* __restrict__ dstI,
                            float* __restrict__ s1, float* __restrict__ s2,
                            float* __restrict__ smn, float* __restrict__ smx, int e_) {
  __shared__ __bf16 sW[FF * EPITCH];   // 80*168*2 = 26.9 KB
  // cooperative vectorized stage of preWt into LDS
  for (int c = threadIdx.x; c < (FF * KEDGE) / 8; c += blockDim.x) {
    int row = (c * 8) / KEDGE, col = (c * 8) % KEDGE;
    *(bf16x8*)&sW[row * EPITCH + col] = *(const bf16x8*)&preWt[row * KEDGE + col];
  }
  __syncthreads();

  int gw = (int)((blockIdx.x * (unsigned)blockDim.x + threadIdx.x) >> 5);
  int lane = threadIdx.x & 31;
  int tb = gw * 16;
  if (tb >= e_) return;
  int r = lane & 15;
  int hl = lane >> 4;
  int ea = tb + r;                    // A-row edge for this lane
  int di = dstI[ea], si = srcI[ea];
  const __bf16* rowD = hbf + (size_t)di * FF;
  const __bf16* rowS = hbf + (size_t)si * FF;

  f32x8 acc[5];
#pragma unroll
  for (int ct = 0; ct < 5; ++ct) acc[ct] = (f32x8){0.f,0.f,0.f,0.f,0.f,0.f,0.f,0.f};

#pragma unroll
  for (int kc = 0; kc < 5; ++kc) {
    int K0 = kc * 32 + hl * 8;        // first 8-run
    int K1 = K0 + 16;                 // second 8-run
    bf16x8 alo = *(const bf16x8*)(K0 < FF ? rowD + K0 : rowS + (K0 - FF));
    bf16x8 ahi = *(const bf16x8*)(K1 < FF ? rowD + K1 : rowS + (K1 - FF));
    bf16x16 a = cat8(alo, ahi);
    int kg = kc * 32 + hl * 8;
#pragma unroll
    for (int ct = 0; ct < 5; ++ct) {
      const __bf16* bp = &sW[(ct * 16 + r) * EPITCH + kg];
      bf16x16 b = cat8(*(const bf16x8*)bp, *(const bf16x8*)(bp + 16));
      acc[ct] = __builtin_amdgcn_wmma_f32_16x16x32_bf16(
          false, a, false, b, (short)0, acc[ct], false, false);
    }
  }

#pragma unroll
  for (int ct = 0; ct < 5; ++ct) {
    int col = ct * 16 + r;
    float bias = preb[col];
#pragma unroll
    for (int i = 0; i < 8; ++i) {
      int M = hl * 8 + i;
      int ee = tb + M;
      if (ee < e_) {
        float m = acc[ct][i] + bias;
        size_t off = (size_t)dstI[ee] * FF + col;
        atomicAdd(s1 + off, m);
        atomicAdd(s2 + off, m * m);
        atomic_min_f32(smn + off, m);
        atomic_max_f32(smx + off, m);
      }
    }
  }
}

// Node GEMM: out[N,80] = X[N,kpad] @ Wt^T + bias.
// Each wave owns a full 16x80 stripe (5 accumulators) so A is loaded once per
// K-chunk. The 80x32 B chunk is staged per-iteration into LDS by the whole
// block (4 waves in lockstep; partial tail handled by clamping, never by
// exiting before a barrier).
__global__ void k_node_gemm(const __bf16* __restrict__ X, const __bf16* __restrict__ Wt,
                            const float* __restrict__ bias, float* __restrict__ out,
                            int nrows, int kpad) {
  __shared__ __bf16 sW[FF * NPITCH];   // 80*40*2 = 6.4 KB
  int lane = threadIdx.x & 31;
  int wib  = threadIdx.x >> 5;        // wave in block (0..3)
  int nrt = (nrows + 15) / 16;
  int rowTile = (int)blockIdx.x * 4 + wib;
  if (rowTile >= nrt) rowTile = nrt - 1;   // clamp: redundant but barrier-safe
  int r = lane & 15, hl = lane >> 4;
  const __bf16* xr = X + (size_t)(rowTile * 16 + r) * kpad;

  f32x8 acc[5];
#pragma unroll
  for (int ct = 0; ct < 5; ++ct) acc[ct] = (f32x8){0.f,0.f,0.f,0.f,0.f,0.f,0.f,0.f};

  int nkc = kpad / 32;
  for (int kc = 0; kc < nkc; ++kc) {
    __syncthreads();   // WAR: previous iteration finished reading sW
    for (int c = threadIdx.x; c < (FF * 32) / 8; c += blockDim.x) {
      int row = (c * 8) / 32, col = (c * 8) % 32;
      *(bf16x8*)&sW[row * NPITCH + col] =
          *(const bf16x8*)&Wt[(size_t)row * kpad + kc * 32 + col];
    }
    __syncthreads();   // RAW: chunk visible to all waves
    int K0 = kc * 32 + hl * 8;
    bf16x16 a = cat8(*(const bf16x8*)(xr + K0), *(const bf16x8*)(xr + K0 + 16));
    int kg = hl * 8;
#pragma unroll
    for (int ct = 0; ct < 5; ++ct) {
      const __bf16* bp = &sW[(ct * 16 + r) * NPITCH + kg];
      bf16x16 b = cat8(*(const bf16x8*)bp, *(const bf16x8*)(bp + 16));
      acc[ct] = __builtin_amdgcn_wmma_f32_16x16x32_bf16(
          false, a, false, b, (short)0, acc[ct], false, false);
    }
  }

#pragma unroll
  for (int ct = 0; ct < 5; ++ct) {
    float bv = bias[ct * 16 + r];
#pragma unroll
    for (int i = 0; i < 8; ++i) {
      int orow = rowTile * 16 + hl * 8 + i;
      if (orow < nrows) out[(size_t)orow * FF + ct * 16 + r] = acc[ct][i] + bv;
    }
  }
}

// ---------------- host orchestration ----------------
extern "C" void kernel_launch(void* const* d_in, const int* in_sizes, int n_in,
                              void* d_out, int out_size, void* d_ws, size_t ws_size,
                              hipStream_t stream) {
  (void)in_sizes; (void)n_in; (void)out_size; (void)ws_size;
  const int*   xfeat = (const int*)d_in[0];
  const int*   eidx  = (const int*)d_in[1];
  const int*   batch = (const int*)d_in[2];
  const float* avg   = (const float*)d_in[3];
  const float* aemb  = (const float*)d_in[4];
  const float* preW  = (const float*)d_in[5];
  const float* preb  = (const float*)d_in[6];
  const float* postW = (const float*)d_in[7];
  const float* postb = (const float*)d_in[8];
  const float* linW  = (const float*)d_in[9];
  const float* linb  = (const float*)d_in[10];
  const float* bng   = (const float*)d_in[11];
  const float* bnb   = (const float*)d_in[12];
  const float* bnm   = (const float*)d_in[13];
  const float* bnv   = (const float*)d_in[14];
  const float* mlpW  = (const float*)d_in[15];
  const float* mlpb  = (const float*)d_in[16];
  float* out = (float*)d_out;

  const int* srcI = eidx;
  const int* dstI = eidx + EE;

  // workspace carve
  char* p = (char*)d_ws;
  auto take = [&](size_t bytes) -> void* {
    void* q = (void*)p;
    p += (bytes + 255) & ~(size_t)255;
    return q;
  };
  float*  h      = (float*) take((size_t)NN * FF * 4);
  __bf16* hbf    = (__bf16*)take((size_t)NN * FF * 2);
  float*  deg    = (float*) take((size_t)NN * 4);
  float*  s1     = (float*) take((size_t)NN * FF * 4);
  float*  s2     = (float*) take((size_t)NN * FF * 4);
  float*  smn    = (float*) take((size_t)NN * FF * 4);
  float*  smx    = (float*) take((size_t)NN * FF * 4);
  __bf16* X      = (__bf16*)take((size_t)NN * KPAD * 2);
  float*  out1   = (float*) take((size_t)NN * FF * 4);
  __bf16* preWt  = (__bf16*)take((size_t)LL * FF * KEDGE * 2);
  __bf16* combWt = (__bf16*)take((size_t)LL * FF * KPAD * 2);
  float*  combB  = (float*) take((size_t)LL * FF * 4);
  float*  gsum   = (float*) take((size_t)NGG * FF * 4);
  float*  gcnt   = (float*) take((size_t)NGG * 4);

  const int T = 256;
  const int nfTiles = (NN * FF + T - 1) / T;          // 15625

  // degree
  k_fill_f32<<<(NN + T - 1) / T, T, 0, stream>>>(deg, NN, 0.f);
  k_deg<<<(EE + T - 1) / T, T, 0, stream>>>(dstI, deg, EE);

  // atom encoder
  k_atom_encoder<<<nfTiles, T, 0, stream>>>(xfeat, aemb, h, hbf, NN);

  // weight prep
  k_prep_preWt<<<(LL * FF * KEDGE + T - 1) / T, T, 0, stream>>>(preW, preWt);
  k_prep_comb<<<(LL * FF * KPAD + T - 1) / T, T, 0, stream>>>(postW, linW, combWt);
  k_prep_combB<<<(LL * FF + T - 1) / T, T, 0, stream>>>(postb, linW, linb, combB);

  // layers
  const int edgeWaves = EE / 16;                        // 50000
  const int edgeBlocks = (edgeWaves * 32 + 127) / 128;  // 4 waves/block
  const int nrt = (NN + 15) / 16;                       // 3125 row tiles
  const int gemmBlocks = (nrt + 3) / 4;                 // 4 row tiles/block
  for (int l = 0; l < LL; ++l) {
    k_init_seg<<<nfTiles, T, 0, stream>>>(s1, s2, smn, smx, NN * FF);
    k_edge_wmma<<<edgeBlocks, 128, 0, stream>>>(
        hbf, preWt + (size_t)l * FF * KEDGE, preb + (size_t)l * FF,
        srcI, dstI, s1, s2, smn, smx, EE);
    k_build_x<<<nfTiles, T, 0, stream>>>(h, deg, avg, s1, s2, smn, smx, X, NN);
    k_node_gemm<<<gemmBlocks, 128, 0, stream>>>(
        X, combWt + (size_t)l * FF * KPAD, combB + (size_t)l * FF, out1, NN, KPAD);
    k_bn_relu_res<<<nfTiles, T, 0, stream>>>(
        out1, bng + (size_t)l * FF, bnb + (size_t)l * FF,
        bnm + (size_t)l * FF, bnv + (size_t)l * FF, h, hbf, NN);
  }

  // pooling + head
  k_fill_f32<<<(NGG * FF + T - 1) / T, T, 0, stream>>>(gsum, NGG * FF, 0.f);
  k_fill_f32<<<(NGG + T - 1) / T, T, 0, stream>>>(gcnt, NGG, 0.f);
  k_pool_accum<<<nfTiles, T, 0, stream>>>(h, batch, gsum, gcnt, NN);
  k_final<<<(NGG + T - 1) / T, T, 0, stream>>>(gsum, gcnt, mlpW, mlpb, out, NGG);
}
